// Conv2d_14147622273082
// MI455X (gfx1250) — compile-verified, matrix operand order checked by reading, not run on hardware
//
#include <hip/hip_runtime.h>
#include <stdint.h>

typedef __bf16 bf16;
typedef __attribute__((ext_vector_type(16))) __bf16 v16bf;
typedef __attribute__((ext_vector_type(8)))  __bf16 v8bf;
typedef __attribute__((ext_vector_type(8)))  float  v8f;

#define CIN   320
#define COUT  320
#define HH    64
#define WW    64
#define BATCH 8
#define KTAPS 9
#define NCHUNK (CIN / 32)

#define BM 128      // output pixels per block (2 h-rows x 64 w)
#define BN 64       // output channels per block
#define BK 32       // cin per WMMA K-step
#define AH 4        // halo rows (h0-1 .. h0+2)
#define AW 66       // halo cols (w=-1 .. 64)
#define ASTR 40     // LDS row stride (bf16): 80B -> conflict-free b128 frag reads
#define BSTR 40

// hf8: 1 sign, 4 exp (bias 14), 3 mantissa. Exactly representable in bf16.
__device__ __forceinline__ float hf8_decode(int bits) {
    int   e    = (bits >> 3) & 0xF;
    float m    = (float)(bits & 7) * 0.125f;
    float mant = (e == 0) ? m : (1.0f + m);
    int   ex   = (e == 0) ? (1 - 14) : (e - 14);
    float v    = ldexpf(mant, ex);
    return (bits & 0x80) ? -v : v;
}

// Pre-pass: w_bits[co][cin][ky][kx] -> wgt_t[tap][co][cin] bf16 (lossless)
__global__ __launch_bounds__(256)
void dequant_weights(const int* __restrict__ w_bits, bf16* __restrict__ wgt_t) {
    int idx = blockIdx.x * blockDim.x + threadIdx.x;
    if (idx >= KTAPS * COUT * CIN) return;
    int tap = idx / (COUT * CIN);
    int rem = idx - tap * (COUT * CIN);
    int co  = rem / CIN;
    int cin = rem - co * CIN;
    wgt_t[idx] = (bf16)hf8_decode(w_bits[(co * CIN + cin) * KTAPS + tap]);
}

// ---- pipeline stage helpers ----
__device__ __forceinline__ void load_a_regs(const float* __restrict__ asrc,
                                            bool avalid, int cin0, float4* areg) {
    if (avalid) {
        #pragma unroll
        for (int j = 0; j < 8; ++j)
            areg[j] = *(const float4*)(asrc + (size_t)(cin0 + j) * (HH * WW));
    } else {
        #pragma unroll
        for (int j = 0; j < 8; ++j) areg[j] = float4{0.f, 0.f, 0.f, 0.f};
    }
}

__device__ __forceinline__ void store_a_lds(bf16* __restrict__ abase, const float4* areg) {
    #pragma unroll
    for (int k = 0; k < 4; ++k) {          // 8x4 register transpose -> 4 contiguous 16B stores
        alignas(16) bf16 t[8];
        #pragma unroll
        for (int j = 0; j < 8; ++j) t[j] = (bf16)(((const float*)&areg[j])[k]);
        *(v8bf*)&abase[k * ASTR] = *(v8bf*)t;
    }
}

__device__ __forceinline__ void issue_b_async(const bf16* __restrict__ g0, uint32_t l0) {
    #pragma unroll
    for (int tap = 0; tap < KTAPS; ++tap) {
        const bf16* g = g0 + (size_t)tap * (COUT * CIN);
        const uint32_t l = l0 + (uint32_t)(tap * (BN * BSTR * 2));
        asm volatile("global_load_async_to_lds_b128 %0, %1, off"
                     :: "v"(l), "v"(g) : "memory");
    }
}

__global__ __launch_bounds__(256)
void conv_wmma(const float* __restrict__ x, const bf16* __restrict__ wgt_t,
               const int* __restrict__ b_bits, float* __restrict__ out) {
    __shared__ bf16 As[2][AH * AW * ASTR];     // 2 x 21.1 KB halo: [hrow][wcol][cin]
    __shared__ bf16 Bs[2][KTAPS * BN * BSTR];  // 2 x 46.1 KB:      [tap][cout][cin]

    const int tid   = threadIdx.x;
    const int mtile = blockIdx.x;              // 0..255
    const int ntile = blockIdx.y;              // 0..4
    const int p0    = mtile * BM;
    const int co0   = ntile * BN;

    const int b  = p0 >> 12;                   // 4096 px / image
    const int h0 = (p0 & 4095) >> 6;           // tile rows h0, h0+1

    // A-tile assignment: (hrow, w-quad, cin-octet) -> 8x global b128 along w
    const int ahrow = tid >> 6;                // 0..3
    const int awq   = (tid >> 2) & 15;         // w quad: w = awq*4 .. +3
    const int ac8   = (tid & 3) * 8;           // cin octet
    const int hin   = h0 - 1 + ahrow;
    const bool avalid = (hin >= 0) && (hin < HH);
    const float* asrc = x + (((size_t)(b * CIN + ac8) * HH + hin) * WW + awq * 4);
    const int aoff = (ahrow * AW + awq * 4 + 1) * ASTR + ac8;

    // B-tile async assignment: one b128 per (thread, tap)
    const int bco  = tid >> 2;                 // 0..63
    const int bseg = (tid & 3) * 8;
    const bf16* bsrc = wgt_t + (size_t)(co0 + bco) * CIN + bseg;
    const int boff = bco * BSTR + bseg;

    // Wave tiling: 8 waves -> 4 (M) x 2 (N); wave = 32x32 = 2x2 WMMA frags
    const int wave  = tid >> 5;
    const int lane  = tid & 31;
    const int wm    = wave & 3;
    const int wn    = wave >> 2;
    const int lrow  = lane & 15;
    const int khalf = (lane >> 4) << 3;        // K base 0/8 per ISA 16-bit layout
    const int r     = wm >> 1;
    const int wo    = (wm & 1) * 32 + lrow;    // + i*16 per frag

    v8f acc[2][2] = {};

    // ---- prologue: halo edge cols (w=-1, w=64) are always zero; zero once ----
    if (tid < 16) {
        const int bufi = tid >> 3, rem = tid & 7;
        const int hr = rem >> 1, side = rem & 1;
        bf16* dst = &As[bufi][(hr * AW + (side ? 65 : 0)) * ASTR];
        v8bf z = {};
        #pragma unroll
        for (int c8 = 0; c8 < 4; ++c8) *(v8bf*)&dst[c8 * 8] = z;
    }
    // stage chunk 0
    {
        float4 areg[8];
        load_a_regs(asrc, avalid, 0, areg);
        issue_b_async(bsrc, (uint32_t)(size_t)&Bs[0][boff]);
        store_a_lds(&As[0][aoff], areg);
    }

    for (int c = 0; c < NCHUNK; ++c) {
        const int cur = c & 1;
        asm volatile("s_wait_asynccnt 0" ::: "memory");   // B(c) landed
        __syncthreads();                                   // A(c)+B(c) visible to all

        float4 areg[8];
        const bool have_next = (c + 1 < NCHUNK);
        if (have_next) {                                   // hide under the WMMA burst
            load_a_regs(asrc, avalid, (c + 1) * BK, areg);
            issue_b_async(bsrc + (size_t)(c + 1) * BK,
                          (uint32_t)(size_t)&Bs[cur ^ 1][boff]);
        }

        // ---- 9 taps x (2x2) WMMA: 36 matrix ops per barrier interval ----
        const bf16* Ab = &As[cur][0];
        const bf16* Bb = &Bs[cur][0];
        #pragma unroll
        for (int tap = 0; tap < KTAPS; ++tap) {
            const int ky = tap / 3;                        // 0..2 == ky_off+1
            const int kx = tap % 3;                        // 0..2 == kx_off+1
            v16bf afrag[2], bfrag[2];
            #pragma unroll
            for (int i = 0; i < 2; ++i) {
                const bf16* ap = &Ab[((r + ky) * AW + wo + i * 16 + kx) * ASTR + khalf];
                *(v8bf*)&afrag[i]       = *(const v8bf*)ap;
                *((v8bf*)&afrag[i] + 1) = *(const v8bf*)(ap + 16);
            }
            #pragma unroll
            for (int j = 0; j < 2; ++j) {
                const bf16* bp = &Bb[(tap * BN + wn * 32 + j * 16 + lrow) * BSTR + khalf];
                *(v8bf*)&bfrag[j]       = *(const v8bf*)bp;
                *((v8bf*)&bfrag[j] + 1) = *(const v8bf*)(bp + 16);
            }
            #pragma unroll
            for (int i = 0; i < 2; ++i)
                #pragma unroll
                for (int j = 0; j < 2; ++j)
                    acc[i][j] = __builtin_amdgcn_wmma_f32_16x16x32_bf16(
                        false, afrag[i], false, bfrag[j],
                        (short)0, acc[i][j], false, false);
        }

        if (have_next)
            store_a_lds(&As[cur ^ 1][aoff], areg);         // cvt after burst; loads landed
    }

    // ---- epilogue: hf8 bias + fp32 store, 32B contiguous per lane ----
    const int mhalf = (lane >> 4) << 3;        // C/D: VGPR v -> M = v (+8 hi lanes)
    #pragma unroll
    for (int j = 0; j < 2; ++j) {
        const int co = co0 + wn * 32 + j * 16 + lrow;
        const float bias = hf8_decode(b_bits[co]);
        float* obase = out + (size_t)(b * COUT + co) * (HH * WW);
        #pragma unroll
        for (int i = 0; i < 2; ++i) {
            const int rowb = (p0 & 4095) + wm * 32 + i * 16 + mhalf;
            float4 o0 = { acc[i][j][0] + bias, acc[i][j][1] + bias,
                          acc[i][j][2] + bias, acc[i][j][3] + bias };
            float4 o1 = { acc[i][j][4] + bias, acc[i][j][5] + bias,
                          acc[i][j][6] + bias, acc[i][j][7] + bias };
            *(float4*)(obase + rowb)     = o0;
            *(float4*)(obase + rowb + 4) = o1;
        }
    }
}

extern "C" void kernel_launch(void* const* d_in, const int* in_sizes, int n_in,
                              void* d_out, int out_size, void* d_ws, size_t ws_size,
                              hipStream_t stream) {
    const float* x      = (const float*)d_in[0];
    const int*   w_bits = (const int*)d_in[1];
    const int*   b_bits = (const int*)d_in[2];
    float*       out    = (float*)d_out;
    bf16*        wgt_t  = (bf16*)d_ws;          // 1.84 MB scratch

    const int total_w = KTAPS * COUT * CIN;
    dequant_weights<<<(total_w + 255) / 256, 256, 0, stream>>>(w_bits, wgt_t);

    dim3 grid((BATCH * HH * WW) / BM, COUT / BN);   // 256 x 5
    conv_wmma<<<grid, 256, 0, stream>>>(x, wgt_t, b_bits, out);
}
